// SelfAttention_45578192945889
// MI455X (gfx1250) — matrix operand order repaired and checked by reading, not compiled, not run on hardware
//
#include <hip/hip_runtime.h>

typedef __bf16 bf16;
typedef __attribute__((ext_vector_type(16))) __bf16 v16bf;
typedef __attribute__((ext_vector_type(8)))  __bf16 v8bf;
typedef __attribute__((ext_vector_type(8)))  float  v8f;
typedef int v4i __attribute__((vector_size(16)));   // matches async-builtin param type

#define SEQ   2048
#define EMB   1024
#define BATCH 4

#define TM 128          // workgroup tile rows
#define TN 128          // workgroup tile cols
#define TK 32           // K chunk per stage
#define LDSW 40         // LDS row stride in elements (32 data + 8 pad, keeps 16B align)

#if __has_builtin(__builtin_amdgcn_global_load_async_to_lds_b128)
#define USE_ASYNC 1
#else
#define USE_ASYNC 0
#endif

typedef __attribute__((address_space(1))) v4i* gvec_t;
typedef __attribute__((address_space(3))) v4i* lvec_t;

__device__ __forceinline__ void wait_async_le(int n) {
#if __has_builtin(__builtin_amdgcn_s_wait_asynccnt)
  if (n == 0) __builtin_amdgcn_s_wait_asynccnt(0);
  else        __builtin_amdgcn_s_wait_asynccnt(4);
#else
  if (n == 0) asm volatile("s_wait_asynccnt 0x0" ::: "memory");
  else        asm volatile("s_wait_asynccnt 0x4" ::: "memory");
#endif
}

// ---------------- elementwise packing kernels ----------------

__global__ __launch_bounds__(256) void pack_bf16_kernel(const float* __restrict__ x,
                                                        bf16* __restrict__ out, long n) {
  long i = (long)blockIdx.x * blockDim.x + threadIdx.x;
  if (i < n) out[i] = (bf16)x[i];
}

// Wt[n][e] = W[e][n], fp32 -> bf16
__global__ __launch_bounds__(256) void transpose_pack_kernel(const float* __restrict__ W,
                                                             bf16* __restrict__ Wt,
                                                             int E, int N) {
  int idx = blockIdx.x * blockDim.x + threadIdx.x;
  if (idx >= E * N) return;
  int n = idx % N;
  int e = idx / N;
  Wt[(long)n * E + e] = (bf16)W[idx];
}

// ---------------- WMMA GEMM: C[M,N] = A[M,K] * Bt[N,K]^T ----------------
// 256 threads (8 waves) compute a 128x128 C tile. Per K-chunk of 32, the A
// slice (128x32) and Bt slice (128x32) are staged into LDS (async double-
// buffered on ASYNCcnt when available). Wave (wr,wc) with wr=wid&3, wc=wid>>2
// computes a 32x64 register tile: 2 A-frags x 4 B-frags -> 8 WMMA accs.
// Fragment layouts follow the ISA 16-bit A 16x32 / B 32x16 VGPR maps.
// OUT_MODE: 0 = bf16 row-major, 1 = bf16 transposed, 2 = fp32 row-major
template <int OUT_MODE>
__global__ __launch_bounds__(256) void gemm_wmma_kernel(
    const bf16* __restrict__ A,  long aBatch, int lda,
    const bf16* __restrict__ Bt, long bBatch, int ldb,
    void* __restrict__ C,        long cBatch, int ldc,
    int Ksz, float scale, int causalSkip, int kClamp) {
  const int tid  = threadIdx.x;
  const int lane = tid & 31;
  const int wid  = tid >> 5;
  const int wr   = wid & 3;    // 32-row group within tile
  const int wc   = wid >> 2;   // 64-col group within tile
  const int n0 = blockIdx.x * TN;
  const int m0 = blockIdx.y * TM;
  const int bz = blockIdx.z;

  // causal: tile fully above the diagonal contributes nothing
  if (causalSkip && n0 > m0 + TM - 1) return;

  int kEnd = Ksz;
  if (kClamp) {  // keys beyond this query tile are exactly zero in P
    int ke = (m0 + TM + TK - 1) & ~(TK - 1);
    kEnd = (ke < Ksz) ? ke : Ksz;
  }

  const bf16* Ab = A  + (long)bz * aBatch;
  const bf16* Bb = Bt + (long)bz * bBatch;

  __shared__ bf16 smA[2][TM * LDSW];
  __shared__ bf16 smB[2][TN * LDSW];

  // Staging map: 512 chunks of 16B per tile slice; thread t moves chunks t and
  // t+256. chunk c -> row c>>2, 16B segment (c&3) of the 64B K-row.
  const int ar0 = tid >> 2,          as0 = (tid & 3) * 8;
  const int ar1 = (tid + 256) >> 2,  as1 = ((tid + 256) & 3) * 8;
  const bf16* gA0 = Ab + (long)(m0 + ar0) * lda + as0;
  const bf16* gA1 = Ab + (long)(m0 + ar1) * lda + as1;
  const bf16* gB0 = Bb + (long)(n0 + ar0) * ldb + as0;
  const bf16* gB1 = Bb + (long)(n0 + ar1) * ldb + as1;
  const int lA0 = ar0 * LDSW + as0, lA1 = ar1 * LDSW + as1;
  const int lB0 = ar0 * LDSW + as0, lB1 = ar1 * LDSW + as1;

#if USE_ASYNC
#define STAGE(buf, k0)                                                          \
  do {                                                                          \
    __builtin_amdgcn_global_load_async_to_lds_b128(                             \
        (gvec_t)(gA0 + (k0)), (lvec_t)&smA[buf][lA0], 0, 0);                    \
    __builtin_amdgcn_global_load_async_to_lds_b128(                             \
        (gvec_t)(gA1 + (k0)), (lvec_t)&smA[buf][lA1], 0, 0);                    \
    __builtin_amdgcn_global_load_async_to_lds_b128(                             \
        (gvec_t)(gB0 + (k0)), (lvec_t)&smB[buf][lB0], 0, 0);                    \
    __builtin_amdgcn_global_load_async_to_lds_b128(                             \
        (gvec_t)(gB1 + (k0)), (lvec_t)&smB[buf][lB1], 0, 0);                    \
  } while (0)
#else
#define STAGE(buf, k0)                                                          \
  do {                                                                          \
    *(v8bf*)&smA[buf][lA0] = *(const v8bf*)(gA0 + (k0));                        \
    *(v8bf*)&smA[buf][lA1] = *(const v8bf*)(gA1 + (k0));                        \
    *(v8bf*)&smB[buf][lB0] = *(const v8bf*)(gB0 + (k0));                        \
    *(v8bf*)&smB[buf][lB1] = *(const v8bf*)(gB1 + (k0));                        \
  } while (0)
#endif

  v8f acc[2][4];
#pragma unroll
  for (int f = 0; f < 2; ++f)
#pragma unroll
    for (int g = 0; g < 4; ++g) acc[f][g] = (v8f){};

  const int lr    = lane & 15;
  const int khalf = (lane >> 4) << 3;   // A frag: 0 or 8
  const int kboff = (lane >> 4) << 3;   // B frag: 0 or 8
  const int arow  = wr * 32;
  const int brow  = wc * 64;

#define COMPUTE(buf)                                                            \
  do {                                                                          \
    v16bf afr[2];                                                               \
    v16bf bfr[4];                                                               \
    _Pragma("unroll") for (int f = 0; f < 2; ++f) {                             \
      int r = (arow + f * 16 + lr) * LDSW;                                      \
      v8bf lo = *(const v8bf*)&smA[buf][r + khalf];                             \
      v8bf hi = *(const v8bf*)&smA[buf][r + khalf + 16];                        \
      afr[f] = __builtin_shufflevector(lo, hi, 0, 1, 2, 3, 4, 5, 6, 7, 8, 9,    \
                                       10, 11, 12, 13, 14, 15);                 \
    }                                                                           \
    _Pragma("unroll") for (int g = 0; g < 4; ++g) {                             \
      int r = (brow + g * 16 + lr) * LDSW;                                      \
      v8bf lo = *(const v8bf*)&smB[buf][r + kboff];                             \
      v8bf hi = *(const v8bf*)&smB[buf][r + kboff + 16];                        \
      bfr[g] = __builtin_shufflevector(lo, hi, 0, 1, 2, 3, 4, 5, 6, 7, 8, 9,    \
                                       10, 11, 12, 13, 14, 15);                 \
    }                                                                           \
    _Pragma("unroll") for (int f = 0; f < 2; ++f)                               \
      _Pragma("unroll") for (int g = 0; g < 4; ++g)                             \
        acc[f][g] = __builtin_amdgcn_wmma_f32_16x16x32_bf16(                    \
            false, afr[f], false, bfr[g], (short)0, acc[f][g], false, false);   \
  } while (0)

#if USE_ASYNC
  STAGE(0, 0);
  int it = 0;
  for (int k0 = 0; k0 < kEnd; k0 += TK, ++it) {
    const int cur = it & 1;
    const bool more = (k0 + TK) < kEnd;
    if (more) STAGE(cur ^ 1, k0 + TK);   // prefetch next chunk into other buffer
    wait_async_le(more ? 4 : 0);         // async loads complete in order
    __syncthreads();
    COMPUTE(cur);
    __syncthreads();
  }
#else
  for (int k0 = 0; k0 < kEnd; k0 += TK) {
    STAGE(0, k0);
    __syncthreads();
    COMPUTE(0);
    __syncthreads();
  }
#endif

  // C/D layout: VGPR r, lanes 0-15: M=+r, N=+lane; lanes 16-31: M=+8+r, N=+lane-16
  const int cm0  = m0 + arow;
  const int cn0  = n0 + brow;
  const int crow = (lane >> 4) << 3;
  const int ccol = lane & 15;
#pragma unroll
  for (int f = 0; f < 2; ++f) {
#pragma unroll
    for (int g = 0; g < 4; ++g) {
#pragma unroll
      for (int r = 0; r < 8; ++r) {
        int m = cm0 + f * 16 + crow + r;
        int n = cn0 + g * 16 + ccol;
        float v = acc[f][g][r] * scale;
        if (OUT_MODE == 0)
          ((bf16*)C)[cBatch * bz + (long)m * ldc + n] = (bf16)v;
        else if (OUT_MODE == 1)
          ((bf16*)C)[cBatch * bz + (long)n * ldc + m] = (bf16)v;
        else
          ((float*)C)[cBatch * bz + (long)m * ldc + n] = v;
      }
    }
  }
#undef STAGE
#undef COMPUTE
}

// ---------------- causal softmax (in place, bf16) ----------------
__global__ __launch_bounds__(256) void softmax_causal_kernel(const bf16* __restrict__ S,
                                                             bf16* __restrict__ P,
                                                             int Ssz) {
  const int q = blockIdx.x;
  const int b = blockIdx.y;
  const long rowoff = ((long)b * Ssz + q) * Ssz;
  const bf16* row = S + rowoff;
  bf16* prow = P + rowoff;
  const int tid = threadIdx.x;
  const int len = q + 1;

  __shared__ float red[256];

  float m = -__builtin_inff();
  for (int j = tid; j < len; j += 256) m = fmaxf(m, (float)row[j]);
  red[tid] = m;
  __syncthreads();
  for (int s = 128; s > 0; s >>= 1) {
    if (tid < s) red[tid] = fmaxf(red[tid], red[tid + s]);
    __syncthreads();
  }
  m = red[0];
  __syncthreads();

  float sum = 0.0f;
  for (int j = tid; j < len; j += 256) sum += __expf((float)row[j] - m);
  red[tid] = sum;
  __syncthreads();
  for (int s = 128; s > 0; s >>= 1) {
    if (tid < s) red[tid] += red[tid + s];
    __syncthreads();
  }
  const float inv = 1.0f / red[0];

  for (int j = tid; j < len; j += 256)
    prow[j] = (bf16)(__expf((float)row[j] - m) * inv);
  for (int j = len + tid; j < Ssz; j += 256)
    prow[j] = (bf16)0.0f;
}

// ---------------- launcher ----------------

extern "C" void kernel_launch(void* const* d_in, const int* in_sizes, int n_in,
                              void* d_out, int out_size, void* d_ws, size_t ws_size,
                              hipStream_t stream) {
  const float* x  = (const float*)d_in[0];
  const float* Wq = (const float*)d_in[1];
  const float* Wk = (const float*)d_in[2];
  const float* Wv = (const float*)d_in[3];
  const float* Wo = (const float*)d_in[4];
  float* out = (float*)d_out;

  char* ws = (char*)d_ws;
  bf16* xb  = (bf16*)ws; ws += (size_t)BATCH * SEQ * EMB * 2;   // 16 MiB
  bf16* Wqt = (bf16*)ws; ws += (size_t)EMB * EMB * 2;           //  2 MiB
  bf16* Wkt = (bf16*)ws; ws += (size_t)EMB * EMB * 2;
  bf16* Wvt = (bf16*)ws; ws += (size_t)EMB * EMB * 2;
  bf16* Wot = (bf16*)ws; ws += (size_t)EMB * EMB * 2;
  bf16* Qb  = (bf16*)ws; ws += (size_t)BATCH * SEQ * EMB * 2;   // 16 MiB
  bf16* Kb  = (bf16*)ws; ws += (size_t)BATCH * SEQ * EMB * 2;   // 16 MiB
  bf16* Vt  = (bf16*)ws; ws += (size_t)BATCH * EMB * SEQ * 2;   // 16 MiB [b][v][s]
  bf16* SP  = (bf16*)ws; ws += (size_t)BATCH * SEQ * SEQ * 2;   // 32 MiB scores->P in place
  bf16* AO  = (bf16*)ws; ws += (size_t)BATCH * SEQ * EMB * 2;   // 16 MiB

  // 1) pack inputs to bf16 (weights transposed so GEMM B operand is K-major)
  {
    long nx = (long)BATCH * SEQ * EMB;
    pack_bf16_kernel<<<dim3((nx + 255) / 256), dim3(256), 0, stream>>>(x, xb, nx);
    int nw = EMB * EMB;
    dim3 gw((nw + 255) / 256), bw(256);
    transpose_pack_kernel<<<gw, bw, 0, stream>>>(Wq, Wqt, EMB, EMB);
    transpose_pack_kernel<<<gw, bw, 0, stream>>>(Wk, Wkt, EMB, EMB);
    transpose_pack_kernel<<<gw, bw, 0, stream>>>(Wv, Wvt, EMB, EMB);
    transpose_pack_kernel<<<gw, bw, 0, stream>>>(Wo, Wot, EMB, EMB);
  }

  const dim3 blk(256);
  const dim3 gProj(EMB / TN, SEQ / TM, BATCH);   // M=SEQ, N=EMB
  const dim3 gScore(SEQ / TN, SEQ / TM, BATCH);  // M=SEQ, N=SEQ

  const long xBat  = (long)SEQ * EMB;
  const long spBat = (long)SEQ * SEQ;
  const long vtBat = (long)EMB * SEQ;

  // 2) projections: Q (with 1/sqrt(dk)=1/32 folded in), K, V^T
  gemm_wmma_kernel<0><<<gProj, blk, 0, stream>>>(xb, xBat, EMB, Wqt, 0, EMB,
                                                 Qb, xBat, EMB, EMB, 0.03125f, 0, 0);
  gemm_wmma_kernel<0><<<gProj, blk, 0, stream>>>(xb, xBat, EMB, Wkt, 0, EMB,
                                                 Kb, xBat, EMB, EMB, 1.0f, 0, 0);
  gemm_wmma_kernel<1><<<gProj, blk, 0, stream>>>(xb, xBat, EMB, Wvt, 0, EMB,
                                                 Vt, vtBat, SEQ, EMB, 1.0f, 0, 0);

  // 3) scores = Q * K^T  (causal tile skip; Bt operand is K itself)
  gemm_wmma_kernel<0><<<gScore, blk, 0, stream>>>(Qb, xBat, EMB, Kb, xBat, EMB,
                                                  SP, spBat, SEQ, EMB, 1.0f, 1, 0);

  // 4) causal softmax in place
  softmax_causal_kernel<<<dim3(SEQ, BATCH), dim3(256), 0, stream>>>(SP, SP, SEQ);

  // 5) AO = P * V  (K-loop clamped to causal extent; Bt operand is V^T)
  gemm_wmma_kernel<0><<<gProj, blk, 0, stream>>>(SP, spBat, SEQ, Vt, vtBat, SEQ,
                                                 AO, xBat, EMB, SEQ, 1.0f, 0, 1);

  // 6) out = AO * Wo  (fp32 store to d_out)
  gemm_wmma_kernel<2><<<gProj, blk, 0, stream>>>(AO, xBat, EMB, Wot, 0, EMB,
                                                 out, xBat, EMB, EMB, 1.0f, 0, 0);

  (void)in_sizes; (void)n_in; (void)out_size; (void)ws_size;
}
